// BiLSTMCRF_80900003987984
// MI455X (gfx1250) — matrix operand order repaired and checked by reading, not compiled, not run on hardware
//
#include <hip/hip_runtime.h>
#include <hip/hip_bf16.h>
#include <math.h>

typedef _Float16 f16;
typedef __attribute__((ext_vector_type(16))) _Float16 v16h;
typedef __attribute__((ext_vector_type(8)))  float    v8f;
typedef __attribute__((ext_vector_type(4)))  int      v4i;

// Problem constants: B=256, S=512, T=16, H=64, E=330 (padded to 352 = 11*32)

// ---- workspace layout (bytes) ----
#define OFF_APACK   0ull
#define SZ_APACK    (8192ull*11*512*2)          /* A fragments of emb: 92.3 MB */
#define OFF_BIH     (OFF_APACK + SZ_APACK)
#define SZ_BIH      (2ull*11*16*512*2)          /* Wih packed B frags [dir][kc][jj][lane][16] */
#define OFF_WHH     (OFF_BIH + SZ_BIH)
#define SZ_WHH      (2ull*16*2*512*2)           /* Whh packed B frags */
#define OFF_WOUT    (OFF_WHH + SZ_WHH)
#define SZ_WOUT     (4ull*512*2)                /* Wout packed B frags */
#define OFF_XPROJ   (OFF_WOUT + SZ_WOUT)
#define SZ_XPROJ    (2ull*8192*16*256*4)        /* x@Wih^T + biases, C-tile layout: 268 MB */
#define OFF_FEATSA  (OFF_XPROJ + SZ_XPROJ)
#define SZ_FEATSA   (8192ull*4*512*2)           /* h (hf||hb) as feats A-frags: 33.5 MB */
#define OFF_FEATS   (OFF_FEATSA + SZ_FEATSA)
#define SZ_FEATS    (8192ull*256*4)             /* emission scores, C-tile layout: 8.4 MB */

static __device__ __forceinline__ v8f wmma16(v16h a, v16h b, v8f c) {
  return __builtin_amdgcn_wmma_f32_16x16x32_f16(false, a, false, b, (short)0, c, false, false);
}

// ---- CDNA5 async global->LDS copy (16B per lane), ASYNCcnt-tracked ----
static __device__ __forceinline__ void async_copy_b128(const void* g, void* l) {
#if __has_builtin(__builtin_amdgcn_global_load_async_to_lds_b128)
  typedef __attribute__((address_space(1))) v4i* gp_t;
  typedef __attribute__((address_space(3))) v4i* lp_t;
  __builtin_amdgcn_global_load_async_to_lds_b128((gp_t)g, (lp_t)l, 0, 0);
#else
  unsigned loff = (unsigned)(unsigned long long)l;   // LDS byte address
  asm volatile("global_load_async_to_lds_b128 %0, %1, off"
               :: "v"(loff), "v"(g) : "memory");
#endif
}

#if __has_builtin(__builtin_amdgcn_s_wait_asynccnt)
#define WAIT_ASYNC(n) __builtin_amdgcn_s_wait_asynccnt(n)
#else
#define WAIT_ASYNC(n) asm volatile("s_wait_asynccnt %0" :: "i"(n) : "memory")
#endif

// -------------------------------------------------------------------------
// Kernel 1: gather 5 embedding tables -> f16 WMMA A-fragment layout
// A layout (16-bit, 16x32): m=lane&15; k = (idx&7) + 16*(idx>>3) + 8*(lane>>4)
// Apack[rtile][kc][lane][16], row = s*256 + b
// -------------------------------------------------------------------------
__global__ void k_embed_pack(const float* __restrict__ we, const float* __restrict__ fe,
                             const float* __restrict__ be, const float* __restrict__ re,
                             const float* __restrict__ pe,
                             const int* __restrict__ wid, const int* __restrict__ fid,
                             const int* __restrict__ bid, const int* __restrict__ rid,
                             const int* __restrict__ pid, f16* __restrict__ apack) {
  long t = (long)blockIdx.x * blockDim.x + threadIdx.x;   // one thread = 8 contiguous-k halves
  const long total = 8192l * 11 * 32 * 2;
  if (t >= total) return;
  int g    = (int)(t & 1);
  int lane = (int)((t >> 1) & 31);
  long rest = t >> 6;
  int kc    = (int)(rest % 11);
  int rtile = (int)(rest / 11);
  int m   = lane & 15;
  int row = rtile * 16 + m;
  int s = row >> 8;
  int b = row & 255;
  int e0 = kc * 32 + g * 16 + 8 * (lane >> 4);  // 8 consecutive e
  int iw = wid[b*512+s], ifl = fid[b*512+s], ib = bid[b*512+s];
  int ir = rid[b*512+s], ip = pid[b*512+s];
  union { f16 h[8]; float4 v4; } out;
  #pragma unroll
  for (int j = 0; j < 8; j++) {
    int e = e0 + j; float v;
    if      (e < 100) v = we[iw*100 + e];
    else if (e < 150) v = fe[ifl*50 + e - 100];
    else if (e < 200) v = be[ib*50  + e - 150];
    else if (e < 250) v = re[ir*50  + e - 200];
    else if (e < 330) v = pe[ip*80  + e - 250];
    else              v = 0.f;
    out.h[j] = (f16)v;
  }
  *reinterpret_cast<float4*>(apack + ((long)(rtile*11 + kc) * 512 + lane*16 + g*8)) = out.v4;
}

// -------------------------------------------------------------------------
// Kernel 2: pack Wih/Whh/Wout into WMMA B-fragment layout
// B layout (16-bit, 32x16): n = lane&15; k = (idx&7) + 16*(idx>>3) + 8*(lane>>4)
// Wih pack: [dir][kc][jj][lane][16]  (kc-contiguous for async staging)
// -------------------------------------------------------------------------
__global__ void k_pack_weights(const float* __restrict__ WihF, const float* __restrict__ WihB,
                               const float* __restrict__ WhhF, const float* __restrict__ WhhB,
                               const float* __restrict__ Wout,
                               f16* __restrict__ bih, f16* __restrict__ whh, f16* __restrict__ wout) {
  int t = blockIdx.x * blockDim.x + threadIdx.x;
  const int N_BIH = 2*11*16*512;   // 180224
  const int N_WHH = 2*16*2*512;    // 32768
  const int N_WOUT = 4*512;        // 2048
  if (t < N_BIH) {
    int idx = t & 15, lane = (t >> 4) & 31;
    int rest = t >> 9;
    int jj = rest & 15; rest >>= 4;
    int kc = rest % 11; int dir = rest / 11;
    int n = jj*16 + (lane & 15);
    int k = kc*32 + (idx & 7) + 16*(idx >> 3) + 8*(lane >> 4);
    const float* W = dir ? WihB : WihF;
    bih[t] = (f16)((k < 330) ? W[n*330 + k] : 0.f);
  } else if (t < N_BIH + N_WHH) {
    int u = t - N_BIH;
    int idx = u & 15, lane = (u >> 4) & 31;
    int rest = u >> 9;
    int kc = rest & 1; rest >>= 1;
    int jj = rest & 15; int dir = rest >> 4;
    int n = jj*16 + (lane & 15);
    int k = kc*32 + (idx & 7) + 16*(idx >> 3) + 8*(lane >> 4);
    const float* W = dir ? WhhB : WhhF;
    whh[u] = (f16)W[n*64 + k];
  } else if (t < N_BIH + N_WHH + N_WOUT) {
    int u = t - N_BIH - N_WHH;
    int idx = u & 15, lane = (u >> 4) & 31;
    int kc = (u >> 9) & 3;
    int n = lane & 15;
    int k = kc*32 + (idx & 7) + 16*(idx >> 3) + 8*(lane >> 4);
    wout[u] = (f16)Wout[n*128 + k];
  }
}

// -------------------------------------------------------------------------
// Kernel 3: input projection GEMM: Xproj = emb @ Wih^T + bih + bhh
// 8 waves/block, one wave per 16-row strip; B slice (16 KB/kc) staged via
// async DMA into a 2x16KB LDS ring, one kc ahead; WMMA B frags from LDS.
// Block-uniform dir (blocks 0..1023 -> fwd, 1024..2047 -> bwd).
// Xproj stored in C-tile layout: [dir][rtile][jj][lane][8] f32
// -------------------------------------------------------------------------
__global__ __launch_bounds__(256) void
k_input_gemm(const f16* __restrict__ apack, const f16* __restrict__ bpack,
             const float* __restrict__ bihF, const float* __restrict__ bhhF,
             const float* __restrict__ bihB, const float* __restrict__ bhhB,
             float* __restrict__ xproj) {
  __shared__ f16 Bs[2][8192];                      // 2 x 16 KB ring
  int wave = threadIdx.x >> 5;
  int lane = threadIdx.x & 31;
  int bdir = blockIdx.x >> 10;                     // uniform over block
  int rtile = ((blockIdx.x & 1023) << 3) + wave;
  const float* b1 = bdir ? bihB : bihF;
  const float* b2 = bdir ? bhhB : bhhF;
  const f16* bsrc = bpack + (long)bdir * 11 * 8192;

  // stage B slice for kc into ring buffer (256 threads x 4 x 16B = 16 KB)
  auto stage = [&](int buf, int kc) {
    const f16* src = bsrc + (long)kc * 8192;
    #pragma unroll
    for (int i = 0; i < 4; i++) {
      int off = threadIdx.x * 8 + i * 2048;        // halves
      async_copy_b128(src + off, &Bs[buf][off]);
    }
  };

  v8f acc[16];
  #pragma unroll
  for (int jj = 0; jj < 16; jj++) {
    float bias = b1[jj*16 + (lane & 15)] + b2[jj*16 + (lane & 15)];
    #pragma unroll
    for (int r = 0; r < 8; r++) acc[jj][r] = bias;
  }

  stage(0, 0);
  for (int kc = 0; kc < 11; kc++) {
    if (kc < 10) {
      stage((kc + 1) & 1, kc + 1);                 // async: overlaps this kc's WMMAs
      __builtin_prefetch(apack + ((long)(rtile*11 + kc + 1) * 512 + lane*16), 0, 1);
      WAIT_ASYNC(4);                               // current kc's 4 copies done; next 4 in flight
    } else {
      WAIT_ASYNC(0);
    }
    __syncthreads();                               // B slice visible to all waves
    v16h a = *reinterpret_cast<const v16h*>(apack + ((long)(rtile*11 + kc) * 512 + lane*16));
    const f16* bb = &Bs[kc & 1][lane * 16];
    #pragma unroll
    for (int jj = 0; jj < 16; jj++) {
      v16h bf = *reinterpret_cast<const v16h*>(bb + jj * 512);   // 2x ds_load_b128
      acc[jj] = wmma16(a, bf, acc[jj]);
    }
    __syncthreads();                               // all waves done reading before overwrite
  }
  #pragma unroll
  for (int jj = 0; jj < 16; jj++)
    *reinterpret_cast<v8f*>(xproj + ((long)((bdir*8192 + rtile)*16 + jj) * 256 + lane*8)) = acc[jj];
}

// -------------------------------------------------------------------------
// Kernel 4: LSTM recurrence. Block = (dir, batch m-tile of 16 rows), 4 waves.
// Wave w owns columns [w*16, w*16+16) -> gate quadruple tiles {w, w+4, w+8, w+12}.
// Whh B-frags pinned in registers; h cycles through LDS (ds_load_b128 A-frags);
// Xproj tile is the WMMA C operand directly; next step's Xproj prefetched.
// -------------------------------------------------------------------------
__global__ __launch_bounds__(128) void
k_lstm(const f16* __restrict__ whh_pack, const float* __restrict__ xproj,
       const float* __restrict__ h0, const float* __restrict__ c0,
       f16* __restrict__ featsA) {
  int dir   = blockIdx.x >> 4;
  int mtile = blockIdx.x & 15;
  int w     = threadIdx.x >> 5;
  int lane  = threadIdx.x & 31;
  __shared__ f16 hstage[16*64];

  for (int t = threadIdx.x; t < 1024; t += 128) {
    int m = t >> 6, h = t & 63;
    hstage[t] = (f16)h0[(dir*256 + mtile*16 + m)*64 + h];
  }
  int col = w*16 + (lane & 15);
  v8f c;
  #pragma unroll
  for (int r = 0; r < 8; r++) {
    int bm = r + 8*(lane >> 4);
    c[r] = c0[(dir*256 + mtile*16 + bm)*64 + col];
  }
  v16h Bh[4][2];
  #pragma unroll
  for (int q = 0; q < 4; q++) {
    int jj = w + q*4;
    #pragma unroll
    for (int kc = 0; kc < 2; kc++)
      Bh[q][kc] = *reinterpret_cast<const v16h*>(
          whh_pack + ((long)((dir*16 + jj)*2 + kc) * 512 + lane*16));
  }
  __syncthreads();

  int m  = lane & 15;
  int kb = 8 * (lane >> 4);
  // feats A-frag scatter coords for this lane's column
  int kg = col + 64*dir;
  int kchunk = kg >> 5;
  int kk = kg & 31;
  int hl = (kk >> 3) & 1;
  int kk2 = kk - 8*hl;
  int idxA = (kk2 & 7) + 8*(kk2 >> 4);

  for (int step = 0; step < 512; step++) {
    int sx = dir ? (511 - step) : step;
    // h_{t-1} -> A fragments: 4x ds_load_b128 (k-halves contiguous per 16B group)
    union { float4 q[2]; v16h v; } A0, A1;
    A0.q[0] = *reinterpret_cast<const float4*>(&hstage[m*64 + kb]);
    A0.q[1] = *reinterpret_cast<const float4*>(&hstage[m*64 + 16 + kb]);
    A1.q[0] = *reinterpret_cast<const float4*>(&hstage[m*64 + 32 + kb]);
    A1.q[1] = *reinterpret_cast<const float4*>(&hstage[m*64 + 48 + kb]);
    int rt = sx*16 + mtile;
    // prefetch next step's Xproj tiles (hides HBM latency on the serial path)
    if (step < 511) {
      int sxn = dir ? (510 - step) : (step + 1);
      int rtn = sxn*16 + mtile;
      #pragma unroll
      for (int q = 0; q < 4; q++)
        __builtin_prefetch(xproj + ((long)((dir*8192 + rtn)*16 + (w + q*4)) * 256 + lane*8), 0, 1);
    }
    v8f z[4];
    #pragma unroll
    for (int q = 0; q < 4; q++) {
      int jj = w + q*4;
      z[q] = *reinterpret_cast<const v8f*>(
          xproj + ((long)((dir*8192 + rt)*16 + jj) * 256 + lane*8));
      z[q] = wmma16(A0.v, Bh[q][0], z[q]);
      z[q] = wmma16(A1.v, Bh[q][1], z[q]);
    }
    v8f hn;
    #pragma unroll
    for (int r = 0; r < 8; r++) {
      float iv = 1.f / (1.f + __expf(-z[0][r]));
      float fv = 1.f / (1.f + __expf(-z[1][r]));
      float gv = tanhf(z[2][r]);
      float ov = 1.f / (1.f + __expf(-z[3][r]));
      c[r] = fv * c[r] + iv * gv;
      hn[r] = ov * tanhf(c[r]);
    }
    __syncthreads();   // everyone done reading old h
    #pragma unroll
    for (int r = 0; r < 8; r++) {
      int bm = r + 8*(lane >> 4);
      f16 hv = (f16)hn[r];
      hstage[bm*64 + col] = hv;
      featsA[((long)(rt*4 + kchunk) * 512) + (bm + 16*hl)*16 + idxA] = hv;
    }
    __syncthreads();   // new h visible
  }
}

// -------------------------------------------------------------------------
// Kernel 5: emission scores: feats = [hf||hb] @ Wout^T + bout (K=128 -> 4 WMMA)
// -------------------------------------------------------------------------
__global__ __launch_bounds__(256) void
k_feats(const f16* __restrict__ featsA, const f16* __restrict__ wout_pack,
        const float* __restrict__ bout, float* __restrict__ feats) {
  int gid  = blockIdx.x * 8 + (threadIdx.x >> 5);
  int lane = threadIdx.x & 31;
  if (gid >= 8192) return;
  v8f acc;
  float bv = bout[lane & 15];
  #pragma unroll
  for (int r = 0; r < 8; r++) acc[r] = bv;
  #pragma unroll
  for (int kc = 0; kc < 4; kc++) {
    v16h a  = *reinterpret_cast<const v16h*>(featsA + ((long)(gid*4 + kc) * 512 + lane*16));
    v16h bw = *reinterpret_cast<const v16h*>(wout_pack + (kc*512 + lane*16));
    acc = wmma16(a, bw, acc);
  }
  *reinterpret_cast<v8f*>(feats + ((long)gid * 256 + lane*8)) = acc;
}

// -------------------------------------------------------------------------
// Kernel 6: CRF forward + real-path score -> loss[b]. 16 lanes = 16 tags.
// -------------------------------------------------------------------------
__global__ __launch_bounds__(128) void
k_crf(const float* __restrict__ feats, const float* __restrict__ trans,
      const int* __restrict__ tags, float* __restrict__ out) {
  __shared__ float tr_s[256];
  for (int t = threadIdx.x; t < 256; t += 128) tr_s[t] = trans[t];
  __syncthreads();
  int lane = threadIdx.x & 31;
  int wv   = threadIdx.x >> 5;
  int b    = blockIdx.x * 8 + wv*2 + (lane >> 4);
  int tag  = lane & 15;
  int base = lane & 16;
  float tr[16];
  #pragma unroll
  for (int p = 0; p < 16; p++) tr[p] = tr_s[tag*16 + p];
  float alpha = (tag == 14) ? 0.f : -10000.f;   // START = 14
  float emit = 0.f, tacc = 0.f;
  for (int s = 0; s < 512; s++) {
    int row = s*256 + b;
    int rt = row >> 4, mm = row & 15;
    float e = feats[((long)rt*32 + tag + ((mm >> 3) << 4)) * 8 + (mm & 7)];
    float vv[16];
    float vmax = -3.4e38f;
    #pragma unroll
    for (int p = 0; p < 16; p++) {
      float ap = __shfl(alpha, base + p, 32);
      vv[p] = ap + tr[p];
      vmax = fmaxf(vmax, vv[p]);
    }
    float ssum = 0.f;
    #pragma unroll
    for (int p = 0; p < 16; p++) ssum += __expf(vv[p] - vmax);
    alpha = e + vmax + __logf(ssum);
    int tg = tags[b*512 + s];
    if (tag == tg) {
      emit += e;
      int pv = s ? tags[b*512 + s - 1] : 14;
      tacc += tr_s[tg*16 + pv];
    }
  }
  float term = alpha + tr_s[15*16 + tag];       // STOP = 15
  float m16 = term;
  #pragma unroll
  for (int off = 8; off > 0; off >>= 1) m16 = fmaxf(m16, __shfl_xor(m16, off, 32));
  float esum = __expf(term - m16);
  #pragma unroll
  for (int off = 8; off > 0; off >>= 1) esum += __shfl_xor(esum, off, 32);
  float all_sc = m16 + __logf(esum);
  #pragma unroll
  for (int off = 8; off > 0; off >>= 1) {
    emit += __shfl_xor(emit, off, 32);
    tacc += __shfl_xor(tacc, off, 32);
  }
  float real = emit + tacc + tr_s[15*16 + tags[b*512 + 511]];
  if (tag == 0) out[b] = all_sc - real;
}

// -------------------------------------------------------------------------
extern "C" void kernel_launch(void* const* d_in, const int* in_sizes, int n_in,
                              void* d_out, int out_size, void* d_ws, size_t ws_size,
                              hipStream_t stream) {
  (void)in_sizes; (void)n_in; (void)out_size; (void)ws_size;
  const float* we   = (const float*)d_in[0];
  const float* fe   = (const float*)d_in[1];
  const float* be   = (const float*)d_in[2];
  const float* re   = (const float*)d_in[3];
  const float* pe   = (const float*)d_in[4];
  const float* WihF = (const float*)d_in[5];
  const float* WhhF = (const float*)d_in[6];
  const float* bihF = (const float*)d_in[7];
  const float* bhhF = (const float*)d_in[8];
  const float* WihB = (const float*)d_in[9];
  const float* WhhB = (const float*)d_in[10];
  const float* bihB = (const float*)d_in[11];
  const float* bhhB = (const float*)d_in[12];
  const float* Wout = (const float*)d_in[13];
  const float* bout = (const float*)d_in[14];
  const float* trans= (const float*)d_in[15];
  const float* h0   = (const float*)d_in[16];
  const float* c0   = (const float*)d_in[17];
  const int* wid    = (const int*)d_in[18];
  const int* fid    = (const int*)d_in[19];
  const int* bid    = (const int*)d_in[20];
  const int* rid    = (const int*)d_in[21];
  const int* pid    = (const int*)d_in[22];
  const int* tags   = (const int*)d_in[23];

  char* ws = (char*)d_ws;
  f16*   apack  = (f16*)(ws + OFF_APACK);
  f16*   bihp   = (f16*)(ws + OFF_BIH);
  f16*   whhp   = (f16*)(ws + OFF_WHH);
  f16*   woutp  = (f16*)(ws + OFF_WOUT);
  float* xproj  = (float*)(ws + OFF_XPROJ);
  f16*   featsA = (f16*)(ws + OFF_FEATSA);
  float* feats  = (float*)(ws + OFF_FEATS);

  k_embed_pack<<<22528, 256, 0, stream>>>(we, fe, be, re, pe, wid, fid, bid, rid, pid, apack);
  k_pack_weights<<<840, 256, 0, stream>>>(WihF, WihB, WhhF, WhhB, Wout, bihp, whhp, woutp);
  k_input_gemm<<<2048, 256, 0, stream>>>(apack, bihp, bihF, bhhF, bihB, bhhB, xproj);
  k_lstm<<<32, 128, 0, stream>>>(whhp, xproj, h0, c0, featsA);
  k_feats<<<1024, 256, 0, stream>>>(featsA, woutp, bout, feats);
  k_crf<<<32, 128, 0, stream>>>(feats, trans, tags, (float*)d_out);
}
